// TDT_Interaction_9216999817555
// MI455X (gfx1250) — matrix operand order repaired and checked by reading, not compiled, not run on hardware
//
#include <hip/hip_runtime.h>
#include <hip/hip_bf16.h>
#include <math.h>

// Problem constants (from reference)
#define N_NODES 40000
#define N_EDGES 640000
#define DIM     128
#define NH      8
#define DHD     16
#define NG      64

typedef __bf16 bf16t;
typedef __attribute__((ext_vector_type(16))) __bf16 v16bf;
typedef __attribute__((ext_vector_type(8)))  float   v8f;

// ---------------------------------------------------------------------------
// WMMA helpers (CDNA5 v_wmma_f32_16x16x32_bf16, wave32)
// ---------------------------------------------------------------------------
__device__ __forceinline__ v8f wmma_bf16(v16bf a, v16bf b, v8f c) {
  return __builtin_amdgcn_wmma_f32_16x16x32_bf16(
      /*neg_a=*/false, a, /*neg_b=*/false, b,
      /*c_mod=*/(short)0, c, /*reuse_a=*/false, /*reuse_b=*/false);
}

// A-matrix 16x32 bf16 fragment. rowp -> start of this lane's row (row = lane&15),
// K contiguous. g = lane>>4. Lane covers K in [8g,8g+8) and [16+8g,16+8g+8).
__device__ __forceinline__ v16bf load_a_frag(const bf16t* rowp, int g) {
  v16bf a;
  const bf16t* p0 = rowp + 8 * g;
  const bf16t* p1 = rowp + 16 + 8 * g;
#pragma unroll
  for (int i = 0; i < 8; ++i) { a[i] = p0[i]; a[8 + i] = p1[i]; }
  return a;
}

// B-matrix 32x16 bf16 fragment from K-contiguous (transposed) weight storage.
// colk -> &Wt[col * Kdim]; lane covers K in [8g,8g+8) and [16+8g,16+8g+8).
__device__ __forceinline__ v16bf load_b_frag(const bf16t* colk, int g) {
  v16bf b;
  const bf16t* p0 = colk + 8 * g;
  const bf16t* p1 = colk + 16 + 8 * g;
#pragma unroll
  for (int i = 0; i < 8; ++i) { b[i] = p0[i]; b[8 + i] = p1[i]; }
  return b;
}

// ---------------------------------------------------------------------------
// Kernel 0: weight convert f32 [K][N] -> bf16 transposed [N][K]
// ---------------------------------------------------------------------------
__global__ __launch_bounds__(256) void wconv_kernel(const float* __restrict__ w,
                                                    bf16t* __restrict__ wt,
                                                    int K, int Nn) {
  int idx = blockIdx.x * blockDim.x + threadIdx.x;
  if (idx >= K * Nn) return;
  int k = idx / Nn, n = idx - k * Nn;
  wt[n * K + k] = (bf16t)w[idx];
}

// ---------------------------------------------------------------------------
// CSR build: zero -> histogram -> single-block scan -> scatter
// ---------------------------------------------------------------------------
__global__ __launch_bounds__(256) void zero_int_kernel(int* __restrict__ p, int n) {
  int i = blockIdx.x * blockDim.x + threadIdx.x;
  if (i < n) p[i] = 0;
}

__global__ __launch_bounds__(256) void hist_kernel(const int* __restrict__ dst,
                                                   int* __restrict__ cnt) {
  int e = blockIdx.x * blockDim.x + threadIdx.x;
  if (e >= N_EDGES) return;
  atomicAdd(&cnt[dst[e]], 1);
}

__global__ __launch_bounds__(1024) void scan_kernel(const int* __restrict__ cnt,
                                                    int* __restrict__ rowstart,
                                                    int* __restrict__ basec) {
  __shared__ int part[1024];
  const int tid = threadIdx.x;
  const int CH = (N_NODES + 1023) / 1024;  // 40 elements per thread
  const int base = tid * CH;
  int s = 0;
  for (int i = 0; i < CH; ++i) {
    int idx = base + i;
    if (idx < N_NODES) s += cnt[idx];
  }
  part[tid] = s;
  __syncthreads();
  for (int off = 1; off < 1024; off <<= 1) {  // inclusive Hillis-Steele
    int other = (tid >= off) ? part[tid - off] : 0;
    __syncthreads();
    part[tid] += other;
    __syncthreads();
  }
  int run = (tid > 0) ? part[tid - 1] : 0;  // exclusive prefix of this chunk
  for (int i = 0; i < CH; ++i) {
    int idx = base + i;
    if (idx < N_NODES) {
      rowstart[idx] = run;
      basec[idx] = run;
      run += cnt[idx];
    }
  }
  if (tid == 1023) rowstart[N_NODES] = run;  // == N_EDGES
}

__global__ __launch_bounds__(256) void scatter_kernel(const int* __restrict__ dst,
                                                      int* __restrict__ basec,
                                                      int* __restrict__ perm) {
  int e = blockIdx.x * blockDim.x + threadIdx.x;
  if (e >= N_EDGES) return;
  int pos = atomicAdd(&basec[dst[e]], 1);
  perm[pos] = e;
}

// ---------------------------------------------------------------------------
// Kernel 1: h = x + t ; q = h@Wq + bq (WMMA)
// One wave per 16-node tile. 40000/16 = 2500 tiles, 4 waves/block -> 625 blocks.
// ---------------------------------------------------------------------------
#define K1_WAVES 4
__global__ __launch_bounds__(32 * K1_WAVES) void node_pre_kernel(
    const float* __restrict__ x, const float* __restrict__ t,
    const bf16t* __restrict__ wt_q, const float* __restrict__ bq,
    float* __restrict__ hbuf, float* __restrict__ qbuf) {
  __shared__ __align__(16) bf16t Hb[K1_WAVES][16][DIM];

  const int wave = threadIdx.x >> 5;
  const int lane = threadIdx.x & 31;
  const int g = lane >> 4;
  const int col = lane & 15;
  const long tile = (long)blockIdx.x * K1_WAVES + wave;
  if (tile * 16 >= N_NODES) return;
  const int n0 = (int)(tile * 16);

  for (int i = lane; i < 16 * DIM; i += 32) {
    int r = i >> 7, c = i & 127;
    long gi = (long)(n0 + r) * DIM + c;
    float hv = x[gi] + t[gi];
    hbuf[gi] = hv;
    Hb[wave][r][c] = (bf16t)hv;
  }
  __builtin_amdgcn_wave_barrier();

  const bf16t* arow = &Hb[wave][lane & 15][0];
  const v16bf a0 = load_a_frag(arow, g);
  const v16bf a1 = load_a_frag(arow + 32, g);
  const v16bf a2 = load_a_frag(arow + 64, g);
  const v16bf a3 = load_a_frag(arow + 96, g);
  for (int n = 0; n < 8; ++n) {
    v8f acc = {};
    const bf16t* bt = wt_q + (n * 16 + col) * DIM;
    acc = wmma_bf16(a0, load_b_frag(bt, g), acc);
    acc = wmma_bf16(a1, load_b_frag(bt + 32, g), acc);
    acc = wmma_bf16(a2, load_b_frag(bt + 64, g), acc);
    acc = wmma_bf16(a3, load_b_frag(bt + 96, g), acc);
    float bias = bq[n * 16 + col];
#pragma unroll
    for (int r = 0; r < 8; ++r)
      qbuf[(long)(n0 + r + 8 * g) * DIM + n * 16 + col] = acc[r] + bias;
  }
}

// ---------------------------------------------------------------------------
// Kernel 2: fused edge pipeline. One wave per 16-edge tile; grid-stride over
// tiles; the three 128x128 weight matrices live in LDS (staged once/block).
//   W  = f@W_lin + b_lin + ssp(f@W_r1 + b_r1)@W_r2 + b_r2, * cutoff(r)
//   m  = h[src] * W                       (bf16 in LDS)
//   k  = m@Wk + bk  -> scores (shfl-reduce over DH=16) -> global
//   v  = m@Wv + bv  -> bf16 to global (staged, 128-bit stores)
// ---------------------------------------------------------------------------
#define K2_WAVES 4
#define K2_GRID  2500
__global__ __launch_bounds__(32 * K2_WAVES) void edge_main_kernel(
    const float* __restrict__ f_ij, const float* __restrict__ r_ij,
    const int* __restrict__ src, const int* __restrict__ dst,
    const float* __restrict__ b_lin, const float* __restrict__ b_r1,
    const float* __restrict__ b_r2, const float* __restrict__ bk,
    const float* __restrict__ bv,
    const bf16t* __restrict__ wt_lin, const bf16t* __restrict__ wt_r1,
    const bf16t* __restrict__ wt_r2, const bf16t* __restrict__ wt_k,
    const bf16t* __restrict__ wt_v,
    const float* __restrict__ hbuf, const float* __restrict__ qbuf,
    float* __restrict__ scores, bf16t* __restrict__ vbuf) {
  __shared__ __align__(16) bf16t Wr2s[DIM * DIM];          // 32KB, [n][k]
  __shared__ __align__(16) bf16t Wks[DIM * DIM];           // 32KB
  __shared__ __align__(16) bf16t Wvs[DIM * DIM];           // 32KB
  __shared__ __align__(16) bf16t Af[K2_WAVES][16][NG];     // f tile (bf16)
  __shared__ __align__(16) bf16t At[K2_WAVES][16][DIM];    // ssp act / V stage
  __shared__ __align__(16) bf16t Am[K2_WAVES][16][DIM];    // messages m (bf16)
  __shared__ int   Sg[K2_WAVES][16];
  __shared__ int   Dg[K2_WAVES][16];
  __shared__ float Cg[K2_WAVES][16];

  const int tid = threadIdx.x;
  const int wave = tid >> 5;
  const int lane = tid & 31;
  const int g = lane >> 4;
  const int col = lane & 15;

  // ---- stage the three KxN=128x128 bf16 weight matrices into LDS ----
  {
    const uint4* s2 = (const uint4*)wt_r2;
    const uint4* sk = (const uint4*)wt_k;
    const uint4* sv = (const uint4*)wt_v;
    uint4* d2 = (uint4*)Wr2s;
    uint4* dk = (uint4*)Wks;
    uint4* dv = (uint4*)Wvs;
    for (int i = tid; i < DIM * DIM * 2 / 16; i += 32 * K2_WAVES) {
      d2[i] = s2[i];
      dk[i] = sk[i];
      dv[i] = sv[i];
    }
  }
  __syncthreads();

  for (long tile = (long)blockIdx.x * K2_WAVES + wave; tile < N_EDGES / 16;
       tile += (long)K2_GRID * K2_WAVES) {
    const int e0 = (int)(tile * 16);

    // prefetch next tile's f rows (16x64 f32 = 4KB = 32 lanes x 128B)
    long ntile = tile + (long)K2_GRID * K2_WAVES;
    if (ntile * 16 < N_EDGES)
      __builtin_prefetch(f_ij + ntile * 16 * NG + lane * 32, 0, 1);

    // edge metadata + cosine cutoff
    if (lane < 16) {
      int e = e0 + lane;
      Sg[wave][lane] = src[e];
      Dg[wave][lane] = dst[e];
      float r = r_ij[e];
      Cg[wave][lane] =
          (r < 8.0f) ? 0.5f * (cosf(0.39269908169872414f * r) + 1.0f) : 0.0f;
    }
    // f tile -> bf16 LDS
    for (int i = lane; i < 16 * NG; i += 32) {
      int r = i >> 6, c = i & 63;
      Af[wave][r][c] = (bf16t)f_ij[(long)(e0 + r) * NG + c];
    }
    __builtin_amdgcn_wave_barrier();

    const bf16t* afr = &Af[wave][lane & 15][0];
    const v16bf af0 = load_a_frag(afr, g);
    const v16bf af1 = load_a_frag(afr + 32, g);

    // --- resnet branch: ssp(f@W_r1 + b_r1) -> At ---
    for (int n = 0; n < 8; ++n) {
      v8f acc = {};
      const bf16t* bt = wt_r1 + (n * 16 + col) * NG;  // L2-resident
      acc = wmma_bf16(af0, load_b_frag(bt, g), acc);
      acc = wmma_bf16(af1, load_b_frag(bt + 32, g), acc);
      float bias = b_r1[n * 16 + col];
#pragma unroll
      for (int r = 0; r < 8; ++r) {
        float u = acc[r] + bias;
        float sp = (u > 20.0f) ? u : log1pf(__expf(u));  // softplus
        At[wave][r + 8 * g][n * 16 + col] = (bf16t)(sp - 0.6931471805599453f);
      }
    }
    __builtin_amdgcn_wave_barrier();

    // --- W = f@W_lin + At@W_r2 + biases, * cutoff; m = h[src]*W -> Am ---
    const bf16t* atr = &At[wave][lane & 15][0];
    const v16bf at0 = load_a_frag(atr, g);
    const v16bf at1 = load_a_frag(atr + 32, g);
    const v16bf at2 = load_a_frag(atr + 64, g);
    const v16bf at3 = load_a_frag(atr + 96, g);
    for (int n = 0; n < 8; ++n) {
      v8f acc = {};
      const bf16t* btl = wt_lin + (n * 16 + col) * NG;  // L2-resident
      acc = wmma_bf16(af0, load_b_frag(btl, g), acc);
      acc = wmma_bf16(af1, load_b_frag(btl + 32, g), acc);
      const bf16t* btr = &Wr2s[(n * 16 + col) * DIM];   // LDS
      acc = wmma_bf16(at0, load_b_frag(btr, g), acc);
      acc = wmma_bf16(at1, load_b_frag(btr + 32, g), acc);
      acc = wmma_bf16(at2, load_b_frag(btr + 64, g), acc);
      acc = wmma_bf16(at3, load_b_frag(btr + 96, g), acc);
      float bias = b_lin[n * 16 + col] + b_r2[n * 16 + col];
#pragma unroll
      for (int r = 0; r < 8; ++r) {
        int row = r + 8 * g;
        float w = (acc[r] + bias) * Cg[wave][row];
        float hv = hbuf[(long)Sg[wave][row] * DIM + n * 16 + col];  // L2 gather
        Am[wave][row][n * 16 + col] = (bf16t)(w * hv);
      }
    }
    __builtin_amdgcn_wave_barrier();

    const bf16t* amr = &Am[wave][lane & 15][0];
    const v16bf am0 = load_a_frag(amr, g);
    const v16bf am1 = load_a_frag(amr + 32, g);
    const v16bf am2 = load_a_frag(amr + 64, g);
    const v16bf am3 = load_a_frag(amr + 96, g);

    // --- k = m@Wk + bk ; scores per (edge, head), head == col-tile n ---
    for (int n = 0; n < 8; ++n) {
      v8f acc = {};
      const bf16t* bt = &Wks[(n * 16 + col) * DIM];     // LDS
      acc = wmma_bf16(am0, load_b_frag(bt, g), acc);
      acc = wmma_bf16(am1, load_b_frag(bt + 32, g), acc);
      acc = wmma_bf16(am2, load_b_frag(bt + 64, g), acc);
      acc = wmma_bf16(am3, load_b_frag(bt + 96, g), acc);
      float bias = bk[n * 16 + col];
#pragma unroll
      for (int r = 0; r < 8; ++r) {
        int row = r + 8 * g;
        float kv = acc[r] + bias;
        float qv = qbuf[(long)Dg[wave][row] * DIM + n * 16 + col];  // L2 gather
        float p = kv * qv;
        p += __shfl_xor(p, 1);
        p += __shfl_xor(p, 2);
        p += __shfl_xor(p, 4);
        p += __shfl_xor(p, 8);  // reduce over 16 lanes of this half-wave
        if (col == 0)
          scores[(long)(e0 + row) * NH + n] = p * 0.25f;  // 1/sqrt(DH)
      }
    }

    // --- v = m@Wv + bv -> stage bf16 into At, then 128-bit global stores ---
    for (int n = 0; n < 8; ++n) {
      v8f acc = {};
      const bf16t* bt = &Wvs[(n * 16 + col) * DIM];     // LDS
      acc = wmma_bf16(am0, load_b_frag(bt, g), acc);
      acc = wmma_bf16(am1, load_b_frag(bt + 32, g), acc);
      acc = wmma_bf16(am2, load_b_frag(bt + 64, g), acc);
      acc = wmma_bf16(am3, load_b_frag(bt + 96, g), acc);
      float bias = bv[n * 16 + col];
#pragma unroll
      for (int r = 0; r < 8; ++r)
        At[wave][r + 8 * g][n * 16 + col] = (bf16t)(acc[r] + bias);
    }
    __builtin_amdgcn_wave_barrier();

    const uint4* vsrc = (const uint4*)&At[wave][0][0];
    uint4* vdst = (uint4*)(vbuf + (long)e0 * DIM);
    for (int i = lane; i < 256; i += 32) vdst[i] = vsrc[i];  // b128 stores
    __builtin_amdgcn_wave_barrier();
  }
}

// ---------------------------------------------------------------------------
// Kernel 3: CSR gather aggregation (no float atomics).
// One wave per node: per-head segment max + denominator (4 lane-replicas per
// head, shfl_xor reduce), then exp-weighted sum of bf16 v rows; each wave
// reads fully coalesced 256B v rows. Writes normalized agg (f32 [N,DIM]).
// ---------------------------------------------------------------------------
#define K3_WAVES 4
__global__ __launch_bounds__(32 * K3_WAVES) void node_gather_kernel(
    const int* __restrict__ rowstart, const int* __restrict__ perm,
    const float* __restrict__ scores, const bf16t* __restrict__ vbuf,
    float* __restrict__ aggn) {
  const int wave = threadIdx.x >> 5;
  const int lane = threadIdx.x & 31;
  const long nd = (long)blockIdx.x * K3_WAVES + wave;
  if (nd >= N_NODES) return;
  const int beg = rowstart[nd];
  const int end = rowstart[nd + 1];

  // phase A: per-head max. lane handles head = lane&7, replica lane>>3 of 4.
  const int head8 = lane & 7;
  float mx = -1e30f;
  for (int i = beg + (lane >> 3); i < end; i += 4)
    mx = fmaxf(mx, scores[(long)perm[i] * NH + head8]);
  mx = fmaxf(mx, __shfl_xor(mx, 8));
  mx = fmaxf(mx, __shfl_xor(mx, 16));

  // phase B: denominator
  float dn = 0.0f;
  for (int i = beg + (lane >> 3); i < end; i += 4)
    dn += __expf(scores[(long)perm[i] * NH + head8] - mx);
  dn += __shfl_xor(dn, 8);
  dn += __shfl_xor(dn, 16);

  // phase C: weighted sum. lane covers cols [4*lane, 4*lane+4), head = lane>>2.
  const int head = lane >> 2;
  const float mxh = __shfl(mx, head);  // lane h (h<8) holds head h's max
  const float dnh = __shfl(dn, head);
  float a0 = 0.0f, a1 = 0.0f, a2 = 0.0f, a3 = 0.0f;
  for (int i = beg; i < end; ++i) {
    int e = perm[i];
    float ex = __expf(scores[(long)e * NH + head] - mxh);
    const bf16t* vp = vbuf + (long)e * DIM + lane * 4;
    a0 += ex * (float)vp[0];
    a1 += ex * (float)vp[1];
    a2 += ex * (float)vp[2];
    a3 += ex * (float)vp[3];
  }
  const float inv = 1.0f / (dnh + 1e-16f);
  float* op = aggn + nd * DIM + lane * 4;
  op[0] = a0 * inv;
  op[1] = a1 * inv;
  op[2] = a2 * inv;
  op[3] = a3 * inv;
}

// ---------------------------------------------------------------------------
// Kernel 4: vout = aggn@Wo + bo; out = LN(x + vout)
// ---------------------------------------------------------------------------
#define K4_WAVES 4
__global__ __launch_bounds__(32 * K4_WAVES) void node_post_kernel(
    const float* __restrict__ x, const float* __restrict__ aggn,
    const bf16t* __restrict__ wt_o, const float* __restrict__ bo,
    const float* __restrict__ ln_g, const float* __restrict__ ln_b,
    float* __restrict__ out) {
  __shared__ __align__(16) bf16t Ab[K4_WAVES][16][DIM];
  __shared__ float Ot[K4_WAVES][16][DIM];

  const int wave = threadIdx.x >> 5;
  const int lane = threadIdx.x & 31;
  const int g = lane >> 4;
  const int col = lane & 15;
  const long tile = (long)blockIdx.x * K4_WAVES + wave;
  if (tile * 16 >= N_NODES) return;
  const int n0 = (int)(tile * 16);

  for (int i = lane; i < 16 * DIM; i += 32) {
    int r = i >> 7, c = i & 127;
    Ab[wave][r][c] = (bf16t)aggn[(long)(n0 + r) * DIM + c];
  }
  __builtin_amdgcn_wave_barrier();

  const bf16t* arow = &Ab[wave][lane & 15][0];
  const v16bf a0 = load_a_frag(arow, g);
  const v16bf a1 = load_a_frag(arow + 32, g);
  const v16bf a2 = load_a_frag(arow + 64, g);
  const v16bf a3 = load_a_frag(arow + 96, g);
  for (int n = 0; n < 8; ++n) {
    v8f acc = {};
    const bf16t* bt = wt_o + (n * 16 + col) * DIM;
    acc = wmma_bf16(a0, load_b_frag(bt, g), acc);
    acc = wmma_bf16(a1, load_b_frag(bt + 32, g), acc);
    acc = wmma_bf16(a2, load_b_frag(bt + 64, g), acc);
    acc = wmma_bf16(a3, load_b_frag(bt + 96, g), acc);
    float bias = bo[n * 16 + col];
#pragma unroll
    for (int r = 0; r < 8; ++r) {
      int row = r + 8 * g;
      Ot[wave][row][n * 16 + col] =
          acc[r] + bias + x[(long)(n0 + row) * DIM + n * 16 + col];
    }
  }
  __builtin_amdgcn_wave_barrier();

  // LayerNorm: row = lane&15, half g covers 64 cols; combine halves via shfl.
  const int row = lane & 15;
  const float* orow = &Ot[wave][row][0];
  float s = 0.0f;
  for (int c = g * 64; c < g * 64 + 64; ++c) s += orow[c];
  s += __shfl_xor(s, 16);
  float mu = s * (1.0f / 128.0f);
  float v = 0.0f;
  for (int c = g * 64; c < g * 64 + 64; ++c) {
    float dv = orow[c] - mu;
    v += dv * dv;
  }
  v += __shfl_xor(v, 16);
  float rstd = rsqrtf(v * (1.0f / 128.0f) + 1e-5f);
  for (int c = g * 64; c < g * 64 + 64; ++c)
    out[(long)(n0 + row) * DIM + c] = (orow[c] - mu) * rstd * ln_g[c] + ln_b[c];
}

// ---------------------------------------------------------------------------
// Host launcher
// ---------------------------------------------------------------------------
extern "C" void kernel_launch(void* const* d_in, const int* in_sizes, int n_in,
                              void* d_out, int out_size, void* d_ws,
                              size_t ws_size, hipStream_t stream) {
  (void)in_sizes; (void)n_in; (void)out_size; (void)ws_size;
  const float* x     = (const float*)d_in[0];
  const float* t     = (const float*)d_in[1];
  const float* f_ij  = (const float*)d_in[2];
  const float* r_ij  = (const float*)d_in[3];
  const int*   src   = (const int*)d_in[4];
  const int*   dst   = (const int*)d_in[5];
  const float* W_lin = (const float*)d_in[6];
  const float* b_lin = (const float*)d_in[7];
  const float* W_r1  = (const float*)d_in[8];
  const float* b_r1  = (const float*)d_in[9];
  const float* W_r2  = (const float*)d_in[10];
  const float* b_r2  = (const float*)d_in[11];
  const float* Wq    = (const float*)d_in[12];
  const float* bq    = (const float*)d_in[13];
  const float* Wk    = (const float*)d_in[14];
  const float* bk    = (const float*)d_in[15];
  const float* Wv    = (const float*)d_in[16];
  const float* bv    = (const float*)d_in[17];
  const float* Wo    = (const float*)d_in[18];
  const float* bo    = (const float*)d_in[19];
  const float* ln_g  = (const float*)d_in[20];
  const float* ln_b  = (const float*)d_in[21];
  float* out = (float*)d_out;

  char* ws = (char*)d_ws;
  size_t off = 0;
  auto alloc = [&](size_t bytes) -> char* {
    char* p = ws + off;
    off += (bytes + 255) & ~(size_t)255;
    return p;
  };
  // ~250 MB workspace total
  bf16t* wt_lin = (bf16t*)alloc((size_t)NG * DIM * 2);
  bf16t* wt_r1  = (bf16t*)alloc((size_t)NG * DIM * 2);
  bf16t* wt_r2  = (bf16t*)alloc((size_t)DIM * DIM * 2);
  bf16t* wt_q   = (bf16t*)alloc((size_t)DIM * DIM * 2);
  bf16t* wt_k   = (bf16t*)alloc((size_t)DIM * DIM * 2);
  bf16t* wt_v   = (bf16t*)alloc((size_t)DIM * DIM * 2);
  bf16t* wt_o   = (bf16t*)alloc((size_t)DIM * DIM * 2);
  float* hbuf   = (float*)alloc((size_t)N_NODES * DIM * 4);
  float* qbuf   = (float*)alloc((size_t)N_NODES * DIM * 4);
  float* aggn   = (float*)alloc((size_t)N_NODES * DIM * 4);
  float* scores = (float*)alloc((size_t)N_EDGES * NH * 4);
  bf16t* vbuf   = (bf16t*)alloc((size_t)N_EDGES * DIM * 2);
  int*   cnt      = (int*)alloc((size_t)N_NODES * 4);
  int*   rowstart = (int*)alloc((size_t)(N_NODES + 1) * 4);
  int*   basec    = (int*)alloc((size_t)N_NODES * 4);
  int*   perm     = (int*)alloc((size_t)N_EDGES * 4);

  // 0) weights -> bf16 transposed [N][K]
  wconv_kernel<<<(NG * DIM + 255) / 256, 256, 0, stream>>>(W_lin, wt_lin, NG, DIM);
  wconv_kernel<<<(NG * DIM + 255) / 256, 256, 0, stream>>>(W_r1, wt_r1, NG, DIM);
  wconv_kernel<<<(DIM * DIM + 255) / 256, 256, 0, stream>>>(W_r2, wt_r2, DIM, DIM);
  wconv_kernel<<<(DIM * DIM + 255) / 256, 256, 0, stream>>>(Wq, wt_q, DIM, DIM);
  wconv_kernel<<<(DIM * DIM + 255) / 256, 256, 0, stream>>>(Wk, wt_k, DIM, DIM);
  wconv_kernel<<<(DIM * DIM + 255) / 256, 256, 0, stream>>>(Wv, wt_v, DIM, DIM);
  wconv_kernel<<<(DIM * DIM + 255) / 256, 256, 0, stream>>>(Wo, wt_o, DIM, DIM);

  // CSR build (histogram -> scan -> scatter)
  zero_int_kernel<<<(N_NODES + 255) / 256, 256, 0, stream>>>(cnt, N_NODES);
  hist_kernel<<<(N_EDGES + 255) / 256, 256, 0, stream>>>(dst, cnt);
  scan_kernel<<<1, 1024, 0, stream>>>(cnt, rowstart, basec);
  scatter_kernel<<<(N_EDGES + 255) / 256, 256, 0, stream>>>(dst, basec, perm);

  // 1) node pre: h, q
  node_pre_kernel<<<(N_NODES / 16 + K1_WAVES - 1) / K1_WAVES, 32 * K1_WAVES, 0,
                    stream>>>(x, t, wt_q, bq, hbuf, qbuf);

  // 2) fused edge pipeline (40000 tiles, grid-stride, LDS-resident weights)
  edge_main_kernel<<<K2_GRID, 32 * K2_WAVES, 0, stream>>>(
      f_ij, r_ij, src, dst, b_lin, b_r1, b_r2, bk, bv, wt_lin, wt_r1, wt_r2,
      wt_k, wt_v, hbuf, qbuf, scores, vbuf);

  // 3) CSR gather aggregation (atomic-free softmax + weighted sum)
  node_gather_kernel<<<(N_NODES + K3_WAVES - 1) / K3_WAVES, 32 * K3_WAVES, 0,
                       stream>>>(rowstart, perm, scores, vbuf, aggn);

  // 4) output projection + residual + LayerNorm
  node_post_kernel<<<(N_NODES / 16 + K4_WAVES - 1) / K4_WAVES, 32 * K4_WAVES, 0,
                     stream>>>(x, aggn, wt_o, bo, ln_g, ln_b, out);
}
